// NonLinearLinearRNNAttn_14104672600490
// MI455X (gfx1250) — compile-verified, matrix operand order checked by reading, not run on hardware
//
#include <hip/hip_runtime.h>
#include <hip/hip_bf16.h>

typedef __attribute__((ext_vector_type(16))) _Float16 v16h;
typedef __attribute__((ext_vector_type(8)))  _Float16 v8h;
typedef __attribute__((ext_vector_type(4)))  _Float16 v4h;
typedef __attribute__((ext_vector_type(8)))  float    v8f;
typedef _Float16 f16;

#define B_ 4
#define L_ 4096
#define D_ 1024
#define H_ 16
#define K_ 64
#define V_ 64
#define C_ 64
#define N_ 64      // L_/C_
#define KW_ 4
#define EPS_ 1e-6f
#define LD_ 72     // padded f16 stride for 64x64 tiles (144B: 16B-aligned, 16-bank spread)

#define V8F0 ((v8f){0.f,0.f,0.f,0.f,0.f,0.f,0.f,0.f})

// ---------------------------------------------------------------------------
// WMMA fragment loader (gfx1250 v_wmma_f32_16x16x32_f16, wave32).
// A lane's 16-bit operand fragment is two contiguous 8-element (16-byte) runs
// along the contraction dim: k in [k0+8*hf, +8) and [k0+16+8*hf, +8).
// Operands are stored outer-dim x k (A row-major, B pre-transposed), so both
// runs lower to ds_load_b128.
// ---------------------------------------------------------------------------
__device__ __forceinline__ v16h ld_frag(const f16* P, int ld, int outer0, int k0,
                                        int lane) {
  const int hf = lane >> 4, idx = lane & 15;
  const f16* base = P + (outer0 + idx) * ld + k0 + (hf << 3);
  v8h lo = *(const v8h*)(base);
  v8h hi = *(const v8h*)(base + 16);
  v16h v;
  #pragma unroll
  for (int e = 0; e < 8; ++e) { v[e] = lo[e]; v[8 + e] = hi[e]; }
  return v;
}

__device__ __forceinline__ v8f wmma16(v16h a, v16h b, v8f c) {
  return __builtin_amdgcn_wmma_f32_16x16x32_f16(false, a, false, b, (short)0, c,
                                                false, false);
}

// D(16x16 tile) += A64[row][k] @ Bt64[col][k]^T   (both stride LD_)
__device__ __forceinline__ v8f mm64(const f16* A, const f16* Bt, int row0,
                                    int col0, int lane, v8f acc) {
  #pragma unroll
  for (int k0 = 0; k0 < 64; k0 += 32)
    acc = wmma16(ld_frag(A, LD_, row0, k0, lane),
                 ld_frag(Bt, LD_, col0, k0, lane), acc);
  return acc;
}

// ---------------------------------------------------------------------------
// Kernel 0a: f32 -> f16 bulk convert (x -> xh), 4 elements/thread
// ---------------------------------------------------------------------------
__global__ __launch_bounds__(256)
void cvt_f16(const float* __restrict__ X, f16* __restrict__ Y, int n4) {
  int i = blockIdx.x * 256 + threadIdx.x;
  if (i < n4) {
    float4 x4 = *(const float4*)&X[(size_t)i * 4];
    v4h y;
    y[0] = (f16)x4.x; y[1] = (f16)x4.y; y[2] = (f16)x4.z; y[3] = (f16)x4.w;
    *(v4h*)&Y[(size_t)i * 4] = y;
  }
}

// ---------------------------------------------------------------------------
// Kernel 0b: weight transpose+convert  WT[n][k] = (f16)W[k][n]
// 64x64 LDS tile; coalesced read over n, coalesced v8h write over k.
// ---------------------------------------------------------------------------
__global__ __launch_bounds__(256)
void transpose_cvt_f16(const float* __restrict__ W, f16* __restrict__ WT,
                       int Kd, int Nw) {
  __shared__ f16 tile[64][72];
  const int tid = threadIdx.x;
  const int n0 = blockIdx.x * 64, k0 = blockIdx.y * 64;
  for (int i = tid; i < 1024; i += 256) {
    int r = i >> 4, c = (i & 15) << 2;  // r: k-local, c: n-local
    float4 w4 = *(const float4*)&W[(size_t)(k0 + r) * Nw + n0 + c];
    tile[c + 0][r] = (f16)w4.x;
    tile[c + 1][r] = (f16)w4.y;
    tile[c + 2][r] = (f16)w4.z;
    tile[c + 3][r] = (f16)w4.w;
  }
  __syncthreads();
  for (int i = tid; i < 512; i += 256) {
    int r = i >> 3, c = (i & 7) << 3;  // r: n-local, c: k-local
    *(v8h*)&WT[(size_t)(n0 + r) * Kd + k0 + c] = *(const v8h*)&tile[r][c];
  }
}

// ---------------------------------------------------------------------------
// Kernel 1/5: all-f16 WMMA GEMM  Out[M,Nw] = Xh[M,Kd] @ WT[Nw,Kd]^T
// 128x128 block tile, 32-wide K steps, double-buffered LDS staging
// (contiguous b128 copies, no conversion), 8 waves x (2x4) 16x16 tiles.
// ---------------------------------------------------------------------------
template <int STORE_HALF>
__global__ __launch_bounds__(256)
void gemm_h16(const f16* __restrict__ Xh, const f16* __restrict__ WT,
              void* __restrict__ Out, int Kd, int Nw) {
  __shared__ f16 sA[2][128 * 40];
  __shared__ f16 sBT[2][128 * 40];
  const int tid = threadIdx.x;
  const int lane = tid & 31, wave = tid >> 5;
  const int wm = wave >> 1, wn = wave & 1;
  const int hf = lane >> 4, lc = lane & 15;
  const int n0 = blockIdx.x * 128;
  const int m0 = blockIdx.y * 128;

  v8f acc[2][4];
  #pragma unroll
  for (int i = 0; i < 2; ++i)
    #pragma unroll
    for (int j = 0; j < 4; ++j) acc[i][j] = V8F0;

  auto stage = [&](int k0, int buf) {
    #pragma unroll
    for (int i8 = tid; i8 < 512; i8 += 256) {   // 128 rows x 32 k, v8h chunks
      int r = i8 >> 2, c = (i8 & 3) << 3;
      *(v8h*)&sA[buf][r * 40 + c]  = *(const v8h*)&Xh[(size_t)(m0 + r) * Kd + k0 + c];
      *(v8h*)&sBT[buf][r * 40 + c] = *(const v8h*)&WT[(size_t)(n0 + r) * Kd + k0 + c];
    }
  };

  stage(0, 0);
  int buf = 0;
  for (int k0 = 0; k0 < Kd; k0 += 32) {
    __syncthreads();
    if (k0 + 32 < Kd) {
      stage(k0 + 32, buf ^ 1);
      __builtin_prefetch(&Xh[(size_t)(m0 + (tid & 127)) * Kd + k0 + 64], 0, 1);
      __builtin_prefetch(&WT[(size_t)(n0 + (tid & 127)) * Kd + k0 + 64], 0, 1);
    }
    v16h af[2], bf[4];
    #pragma unroll
    for (int sm = 0; sm < 2; ++sm) af[sm] = ld_frag(sA[buf], 40, wm * 32 + sm * 16, 0, lane);
    #pragma unroll
    for (int sn = 0; sn < 4; ++sn) bf[sn] = ld_frag(sBT[buf], 40, wn * 64 + sn * 16, 0, lane);
    #pragma unroll
    for (int sm = 0; sm < 2; ++sm)
      #pragma unroll
      for (int sn = 0; sn < 4; ++sn)
        acc[sm][sn] = wmma16(af[sm], bf[sn], acc[sm][sn]);
    buf ^= 1;
  }
  #pragma unroll
  for (int sm = 0; sm < 2; ++sm)
    #pragma unroll
    for (int sn = 0; sn < 4; ++sn) {
      int rb = m0 + wm * 32 + sm * 16;
      int cb = n0 + wn * 64 + sn * 16 + lc;
      #pragma unroll
      for (int g = 0; g < 8; ++g) {
        size_t idx = (size_t)(rb + g + 8 * hf) * Nw + cb;
        if (STORE_HALF) ((f16*)Out)[idx] = (f16)acc[sm][sn][g];
        else            ((float*)Out)[idx] = acc[sm][sn][g];
      }
    }
}

// ---------------------------------------------------------------------------
// Kernel 2: causal depthwise conv(KW=4)+SiLU, per-head L2 norm, head split,
// beta (sigmoid) and gk (decay) projections. One block per (b,l).
// ---------------------------------------------------------------------------
__global__ __launch_bounds__(256)
void conv_norm_proj(const float* __restrict__ x,
                    const f16* __restrict__ Yq, const f16* __restrict__ Yk,
                    const f16* __restrict__ Yv,
                    const float* __restrict__ qcw, const float* __restrict__ kcw,
                    const float* __restrict__ vcw,
                    const float* __restrict__ bw, const float* __restrict__ bb,
                    const float* __restrict__ gw, const float* __restrict__ gb,
                    const float* __restrict__ A_log, const float* __restrict__ dt_bias,
                    f16* __restrict__ qg, f16* __restrict__ kg, f16* __restrict__ vg,
                    float* __restrict__ betag, float* __restrict__ gkg) {
  __shared__ float qs[1024], ks[1024], vs[1024];
  __shared__ float qn[16], kn[16];
  __shared__ float bred[128], gred[128];
  const int tid = threadIdx.x;
  const int b = blockIdx.x / L_, l = blockIdx.x % L_;
  const int c4 = tid << 2;  // 4 consecutive channels per thread

  float aq[4] = {0.f, 0.f, 0.f, 0.f};
  float ak[4] = {0.f, 0.f, 0.f, 0.f};
  float av[4] = {0.f, 0.f, 0.f, 0.f};
  #pragma unroll
  for (int j = 0; j < KW_; ++j) {
    int t = l - (KW_ - 1) + j;
    if (t >= 0) {
      size_t yi = ((size_t)b * L_ + t) * 1024 + c4;
      v4h q4 = *(const v4h*)&Yq[yi];
      v4h k4 = *(const v4h*)&Yk[yi];
      v4h v4 = *(const v4h*)&Yv[yi];
      #pragma unroll
      for (int e = 0; e < 4; ++e) {
        aq[e] += (float)q4[e] * qcw[(c4 + e) * KW_ + j];
        ak[e] += (float)k4[e] * kcw[(c4 + e) * KW_ + j];
        av[e] += (float)v4[e] * vcw[(c4 + e) * KW_ + j];
      }
    }
  }
  #pragma unroll
  for (int e = 0; e < 4; ++e) {
    qs[c4 + e] = aq[e] / (1.f + __expf(-aq[e]));
    ks[c4 + e] = ak[e] / (1.f + __expf(-ak[e]));
    vs[c4 + e] = av[e] / (1.f + __expf(-av[e]));
  }

  // beta / gk projections: 8 threads per head, 128-element partials
  {
    const float* xr = x + ((size_t)b * L_ + l) * D_;
    int h = (tid & 127) >> 3, part = tid & 7;
    const float* wp = (tid < 128) ? bw : gw;
    float s = 0.f;
    for (int d = part * 128; d < part * 128 + 128; ++d) s += xr[d] * wp[d * H_ + h];
    if (tid < 128) bred[tid] = s; else gred[tid - 128] = s;
  }
  __syncthreads();

  if (tid < 16) {
    float s = 0.f;
    #pragma unroll
    for (int e = 0; e < 64; ++e) { float v = qs[tid * 64 + e]; s += v * v; }
    qn[tid] = sqrtf(s);
  } else if (tid < 32) {
    int h = tid - 16; float s = 0.f;
    #pragma unroll
    for (int e = 0; e < 64; ++e) { float v = ks[h * 64 + e]; s += v * v; }
    kn[h] = sqrtf(s);
  } else if (tid < 48) {
    int h = tid - 32; float s = bb[h];
    #pragma unroll
    for (int p = 0; p < 8; ++p) s += bred[h * 8 + p];
    betag[((size_t)b * H_ + h) * L_ + l] = 1.f / (1.f + __expf(-s));
  } else if (tid < 64) {
    int h = tid - 48; float z = gb[h] + dt_bias[h];
    #pragma unroll
    for (int p = 0; p < 8; ++p) z += gred[h * 8 + p];
    float sp = (z > 20.f) ? z : log1pf(__expf(z));
    gkg[((size_t)b * H_ + h) * L_ + l] = -__expf(A_log[h]) * sp;
  }
  __syncthreads();

  {
    int h = c4 >> 6;
    float qninv = 1.f / (qn[h] + EPS_), kninv = 1.f / (kn[h] + EPS_);
    size_t oi = (((size_t)b * H_ + h) * L_ + l) * 64 + (c4 & 63);
    v4h q4, k4, v4;
    #pragma unroll
    for (int e = 0; e < 4; ++e) {
      q4[e] = (f16)(qs[c4 + e] * qninv);
      k4[e] = (f16)(ks[c4 + e] * kninv);
      v4[e] = (f16)vs[c4 + e];
    }
    *(v4h*)&qg[oi] = q4;
    *(v4h*)&kg[oi] = k4;
    *(v4h*)&vg[oi] = v4;
  }
}

// ---------------------------------------------------------------------------
// Kernel 3: chunk-parallel WY transform. One block per (b,h,chunk).
// M = Kbu @ Kdn^T (WMMA), T = (I+tril(M,-1))^{-1} (forward subst, VALU),
// Wt = T@Kbu, Ut = T@Vb, A = tril(Qu @ Kdn^T)  (all WMMA, f16 out).
// LDS: A-operands row-major [row][k]; B-operands transposed [col][k].
// ---------------------------------------------------------------------------
#define TILE_B 9216   // 64*LD_*2
#define WY_LDS 88832
__global__ __launch_bounds__(256)
void wy_chunk(const f16* __restrict__ qg, const f16* __restrict__ kg,
              const f16* __restrict__ vg,
              const float* __restrict__ betag, const float* __restrict__ gkg,
              f16* __restrict__ Wtg, f16* __restrict__ Utg, f16* __restrict__ Ag) {
  extern __shared__ char smem[];
  f16* sKbu  = (f16*)(smem);              // [c][k]  A for M
  f16* sKbuT = (f16*)(smem + 1 * TILE_B); // [k][c]  B for Wt
  f16* sKdn  = (f16*)(smem + 2 * TILE_B); // [c][k]  B for (.)@Kdn^T
  f16* sVbT  = (f16*)(smem + 3 * TILE_B); // [v][c]  B for Ut
  f16* sQu   = (f16*)(smem + 4 * TILE_B); // [c][k]  A
  f16* sTh   = (f16*)(smem + 5 * TILE_B); // [c][d]  A
  float* sM  = (float*)(smem + 55296);
  float* sT  = (float*)(smem + 71680);
  float* bbv = (float*)(smem + 88064);
  float* rrv = (float*)(smem + 88320);
  float* riv = (float*)(smem + 88576);

  const int tid = threadIdx.x, lane = tid & 31, wave = tid >> 5;
  const int hf = lane >> 4, lc = lane & 15;
  const int n  = blockIdx.x % N_;
  const int bh = blockIdx.x / N_;
  const size_t lbase = (size_t)bh * L_ + (size_t)n * C_;

  if (tid < 64) {
    bbv[tid] = betag[lbase + tid];
    float s = 0.f;
    for (int m = 0; m <= tid; ++m) s += gkg[lbase + m];   // in-chunk cumsum
    rrv[tid] = __expf(fmaxf(s, -30.f));
    riv[tid] = __expf(fminf(-s, 30.f));
  }
  __syncthreads();
  for (int i8 = tid; i8 < 512; i8 += 256) {     // 4096 elements / 8
    int r = i8 >> 3, c = (i8 & 7) << 3;
    v8h q8 = *(const v8h*)&qg[lbase * 64 + r * 64 + c];
    v8h k8 = *(const v8h*)&kg[lbase * 64 + r * 64 + c];
    v8h v8 = *(const v8h*)&vg[lbase * 64 + r * 64 + c];
    float br = bbv[r], rr = rrv[r], ri = riv[r];
    v8h qu, kb, kd;
    #pragma unroll
    for (int e = 0; e < 8; ++e) {
      float qv = (float)q8[e], kv = (float)k8[e], vv = (float)v8[e];
      qu[e] = (f16)(rr * qv);
      kb[e] = (f16)(br * rr * kv);
      kd[e] = (f16)(ri * kv);
      sKbuT[(c + e) * LD_ + r] = kb[e];
      sVbT[(c + e) * LD_ + r] = (f16)(br * vv);
    }
    *(v8h*)&sQu[r * LD_ + c]  = qu;
    *(v8h*)&sKbu[r * LD_ + c] = kb;
    *(v8h*)&sKdn[r * LD_ + c] = kd;
  }
  __syncthreads();
  // M = Kbu @ Kdn^T
  #pragma unroll
  for (int t8 = 0; t8 < 2; ++t8) {
    int t = wave * 2 + t8, r0 = (t >> 2) * 16, c0 = (t & 3) * 16;
    v8f a = mm64(sKbu, sKdn, r0, c0, lane, V8F0);
    #pragma unroll
    for (int g = 0; g < 8; ++g) sM[(r0 + g + 8 * hf) * 64 + c0 + lc] = a[g];
  }
  __syncthreads();
  // T = (I + strict_lower(M))^{-1}: unit-lower forward substitution
  for (int i = tid; i < 4096; i += 256) sT[i] = ((i >> 6) == (i & 63)) ? 1.f : 0.f;
  __syncthreads();
  for (int i = 1; i < 64; ++i) {
    if (tid < i) {
      float s = 0.f;
      for (int m = tid; m < i; ++m) s += sM[i * 64 + m] * sT[m * 64 + tid];
      sT[i * 64 + tid] = -s;
    }
    __syncthreads();
  }
  for (int i = tid; i < 4096; i += 256) sTh[(i >> 6) * LD_ + (i & 63)] = (f16)sT[i];
  __syncthreads();
  // Wt = T@Kbu, Ut = T@Vb, A = tril(Qu @ Kdn^T)  -> global f16
  #pragma unroll
  for (int t8 = 0; t8 < 2; ++t8) {
    int t = wave * 2 + t8, r0 = (t >> 2) * 16, c0 = (t & 3) * 16;
    v8f aw = mm64(sTh, sKbuT, r0, c0, lane, V8F0);
    v8f au = mm64(sTh, sVbT,  r0, c0, lane, V8F0);
    v8f aa = mm64(sQu, sKdn,  r0, c0, lane, V8F0);
    #pragma unroll
    for (int g = 0; g < 8; ++g) {
      int rr = r0 + g + 8 * hf, cc = c0 + lc;
      Wtg[(lbase + rr) * 64 + cc] = (f16)aw[g];
      Utg[(lbase + rr) * 64 + cc] = (f16)au[g];
      Ag[(((size_t)bh * N_ + n) * C_ + rr) * C_ + cc] = (rr >= cc) ? (f16)aa[g] : (f16)0.f;
    }
  }
}

// ---------------------------------------------------------------------------
// Kernel 4: sequential state scan. One block per (b,h), 64 chunks.
// Ui = Ut - Wt@S ; o = Qu@S + A@Ui ; S = rmsnorm(r_tot*(S + Kdn^T@Ui))
// State: f32 row-major [k][v] + f16 transposed mirror ShT [v][k] for WMMA.
// Output o written f16 so the output projection consumes it natively.
// ---------------------------------------------------------------------------
#define SCAN_LDS 98816
__global__ __launch_bounds__(256)
void scan_state(const f16* __restrict__ qg, const f16* __restrict__ kg,
                const f16* __restrict__ Wtg, const f16* __restrict__ Utg,
                const f16* __restrict__ Ag,
                const float* __restrict__ gkg, const float* __restrict__ S0,
                const float* __restrict__ rmsw, f16* __restrict__ ogh) {
  extern __shared__ char smem[];
  f16* sQu   = (f16*)(smem);              // [c][k]  A
  f16* sKdnT = (f16*)(smem + 1 * TILE_B); // [k][c]  A (as Kdn^T)
  f16* sWt   = (f16*)(smem + 2 * TILE_B); // [c][k]  A
  f16* sUt   = (f16*)(smem + 3 * TILE_B); // [c][v]  elementwise
  f16* sA    = (f16*)(smem + 4 * TILE_B); // [c][d]  A
  f16* sShT  = (f16*)(smem + 5 * TILE_B); // [v][k]  B (as S)
  f16* sUiT  = (f16*)(smem + 6 * TILE_B); // [v][c]  B (as Ui)
  float* sS   = (float*)(smem + 64512);
  float* sTmp = (float*)(smem + 80896);
  float* red  = (float*)(smem + 97280);
  float* rrv  = (float*)(smem + 98304);
  float* riv  = (float*)(smem + 98560);

  const int tid = threadIdx.x, lane = tid & 31, wave = tid >> 5;
  const int hf = lane >> 4, lc = lane & 15;
  const int bh = blockIdx.x;
  const int b = bh / H_, h = bh % H_;
  const float rw = rmsw[h];

  for (int i = tid; i < 4096; i += 256) {
    float v = S0[(size_t)bh * 4096 + i];
    sS[i] = v;
    sShT[(i & 63) * LD_ + (i >> 6)] = (f16)v;
  }

  for (int n = 0; n < N_; ++n) {
    const size_t lbase = (size_t)bh * L_ + (size_t)n * C_;
    __syncthreads();
    if (tid < 64) {
      float s = 0.f;
      for (int m = 0; m <= tid; ++m) s += gkg[lbase + m];
      rrv[tid] = __expf(fmaxf(s, -30.f));
      riv[tid] = __expf(fminf(-s, 30.f));
    }
    __syncthreads();
    for (int i8 = tid; i8 < 512; i8 += 256) {
      int r = i8 >> 3, c = (i8 & 7) << 3;
      v8h q8 = *(const v8h*)&qg[lbase * 64 + r * 64 + c];
      v8h k8 = *(const v8h*)&kg[lbase * 64 + r * 64 + c];
      v8h w8 = *(const v8h*)&Wtg[lbase * 64 + r * 64 + c];
      v8h u8 = *(const v8h*)&Utg[lbase * 64 + r * 64 + c];
      v8h a8 = *(const v8h*)&Ag[((size_t)bh * N_ + n) * 4096 + r * 64 + c];
      float rr = rrv[r], ri = riv[r];
      v8h qu;
      #pragma unroll
      for (int e = 0; e < 8; ++e) {
        qu[e] = (f16)(rr * (float)q8[e]);
        sKdnT[(c + e) * LD_ + r] = (f16)(ri * (float)k8[e]);
      }
      *(v8h*)&sQu[r * LD_ + c] = qu;
      *(v8h*)&sWt[r * LD_ + c] = w8;
      *(v8h*)&sUt[r * LD_ + c] = u8;
      *(v8h*)&sA[r * LD_ + c]  = a8;
    }
    __syncthreads();
    // Ui = Ut - Wt @ S   (stored transposed for B-operand reuse)
    #pragma unroll
    for (int t8 = 0; t8 < 2; ++t8) {
      int t = wave * 2 + t8, r0 = (t >> 2) * 16, c0 = (t & 3) * 16;
      v8f a = mm64(sWt, sShT, r0, c0, lane, V8F0);
      #pragma unroll
      for (int g = 0; g < 8; ++g) {
        int rr = r0 + g + 8 * hf, cc = c0 + lc;
        sUiT[cc * LD_ + rr] = (f16)((float)sUt[rr * LD_ + cc] - a[g]);
      }
    }
    __syncthreads();
    // o = Qu@S + A@Ui  -> global (B,L,H,V) f16
    #pragma unroll
    for (int t8 = 0; t8 < 2; ++t8) {
      int t = wave * 2 + t8, r0 = (t >> 2) * 16, c0 = (t & 3) * 16;
      v8f a = mm64(sQu, sShT, r0, c0, lane, V8F0);
      a = mm64(sA, sUiT, r0, c0, lane, a);
      #pragma unroll
      for (int g = 0; g < 8; ++g) {
        int rr = r0 + g + 8 * hf;
        int l = n * C_ + rr;
        ogh[(((size_t)b * L_ + l) * H_ + h) * V_ + c0 + lc] = (f16)a[g];
      }
    }
    // S_new = r_tot * (S + Kdn^T @ Ui)
    float rt = rrv[63];
    #pragma unroll
    for (int t8 = 0; t8 < 2; ++t8) {
      int t = wave * 2 + t8, r0 = (t >> 2) * 16, c0 = (t & 3) * 16;
      v8f a = mm64(sKdnT, sUiT, r0, c0, lane, V8F0);
      #pragma unroll
      for (int g = 0; g < 8; ++g) {
        int idx = (r0 + g + 8 * hf) * 64 + c0 + lc;
        sTmp[idx] = rt * (sS[idx] + a[g]);
      }
    }
    __syncthreads();
    // RMS-norm of the 64x64 state
    float ps = 0.f;
    for (int i = tid; i < 4096; i += 256) { float v = sTmp[i]; ps += v * v; }
    red[tid] = ps;
    __syncthreads();
    if (tid == 0) {
      float s = 0.f;
      for (int i = 0; i < 256; ++i) s += red[i];
      red[0] = sqrtf(fmaxf(s * (1.f / 4096.f), EPS_));
    }
    __syncthreads();
    float rms = red[0];
    for (int i = tid; i < 4096; i += 256) {
      float v = sTmp[i] / rms * rw;
      sS[i] = v;
      sShT[(i & 63) * LD_ + (i >> 6)] = (f16)v;
    }
  }
}

// ---------------------------------------------------------------------------
extern "C" void kernel_launch(void* const* d_in, const int* in_sizes, int n_in,
                              void* d_out, int out_size, void* d_ws, size_t ws_size,
                              hipStream_t stream) {
  const float* x       = (const float*)d_in[0];
  const float* S0      = (const float*)d_in[1];
  const float* Wq      = (const float*)d_in[2];
  const float* Wk      = (const float*)d_in[3];
  const float* Wv      = (const float*)d_in[4];
  const float* Wo      = (const float*)d_in[5];
  const float* beta_w  = (const float*)d_in[6];
  const float* beta_b  = (const float*)d_in[7];
  const float* gk_w    = (const float*)d_in[8];
  const float* gk_b    = (const float*)d_in[9];
  const float* qcw     = (const float*)d_in[10];
  const float* kcw     = (const float*)d_in[11];
  const float* vcw     = (const float*)d_in[12];
  const float* rmsw    = (const float*)d_in[13];
  const float* A_log   = (const float*)d_in[14];
  const float* dt_bias = (const float*)d_in[15];
  (void)in_sizes; (void)n_in; (void)out_size; (void)ws_size;

  char* ws = (char*)d_ws;
  size_t off = 0;
  auto take = [&](size_t bytes) {
    char* p = ws + off;
    off += (bytes + 255) & ~(size_t)255;
    return p;
  };
  const size_t BL = (size_t)B_ * L_;
  f16*  Yq    = (f16*)take(BL * 1024 * sizeof(f16));   // reused as Wt after conv
  f16*  Yk    = (f16*)take(BL * 1024 * sizeof(f16));   // reused as Ut
  f16*  Yv    = (f16*)take(BL * 1024 * sizeof(f16));   // reused as A
  f16*  qg    = (f16*)take(BL * 1024 * sizeof(f16));   // (B,H,L,K)
  f16*  kg    = (f16*)take(BL * 1024 * sizeof(f16));
  f16*  vg    = (f16*)take(BL * 1024 * sizeof(f16));
  float* betag = (float*)take(BL * H_ * sizeof(float)); // (B,H,L)
  float* gkg   = (float*)take(BL * H_ * sizeof(float));
  f16*  xh    = (f16*)take(BL * 1024 * sizeof(f16));   // f16 copy of x
  f16*  WqT   = (f16*)take((size_t)D_ * 1024 * sizeof(f16)); // transposed f16 weights
  f16*  WkT   = (f16*)take((size_t)D_ * 1024 * sizeof(f16));
  f16*  WvT   = (f16*)take((size_t)D_ * 1024 * sizeof(f16));
  f16*  WoT   = (f16*)take((size_t)D_ * 1024 * sizeof(f16));
  f16*  ogh   = (f16*)take(BL * 1024 * sizeof(f16));   // (B,L,H*V) f16

  dim3 blk(256);
  dim3 ggrid(1024 / 128, (unsigned)(BL / 128));  // (8, 128)
  dim3 tgrid(1024 / 64, 1024 / 64);              // weight transpose tiles

  // 0) one-time f16 conversion / weight transposition
  cvt_f16<<<dim3((unsigned)(BL * 1024 / 4 / 256)), blk, 0, stream>>>(x, xh, (int)(BL * 1024 / 4));
  transpose_cvt_f16<<<tgrid, blk, 0, stream>>>(Wq, WqT, D_, H_ * K_);
  transpose_cvt_f16<<<tgrid, blk, 0, stream>>>(Wk, WkT, D_, H_ * K_);
  transpose_cvt_f16<<<tgrid, blk, 0, stream>>>(Wv, WvT, D_, H_ * V_);
  transpose_cvt_f16<<<tgrid, blk, 0, stream>>>(Wo, WoT, H_ * V_, D_);

  // 1) projections (all-f16 WMMA GEMMs, double-buffered staging)
  gemm_h16<1><<<ggrid, blk, 0, stream>>>(xh, WqT, Yq, D_, H_ * K_);
  gemm_h16<1><<<ggrid, blk, 0, stream>>>(xh, WkT, Yk, D_, H_ * K_);
  gemm_h16<1><<<ggrid, blk, 0, stream>>>(xh, WvT, Yv, D_, H_ * V_);

  // 2) conv + silu + l2norm + beta/gk
  conv_norm_proj<<<dim3((unsigned)BL), blk, 0, stream>>>(
      x, Yq, Yk, Yv, qcw, kcw, vcw, beta_w, beta_b, gk_w, gk_b, A_log, dt_bias,
      qg, kg, vg, betag, gkg);

  // 3) chunk-parallel WY transform (Wt/Ut/A overlay the dead Yq/Yk/Yv)
  f16* Wtg = Yq;
  f16* Utg = Yk;
  f16* Ag  = Yv;
  wy_chunk<<<dim3(B_ * H_ * N_), blk, WY_LDS, stream>>>(
      qg, kg, vg, betag, gkg, Wtg, Utg, Ag);

  // 4) sequential state scan per (b,h), f16 output
  scan_state<<<dim3(B_ * H_), blk, SCAN_LDS, stream>>>(
      qg, kg, Wtg, Utg, Ag, gkg, S0, rmsw, ogh);

  // 5) output projection -> f32 d_out
  gemm_h16<0><<<ggrid, blk, 0, stream>>>(ogh, WoT, (float*)d_out, H_ * V_, D_);
}
